// GENELink_4475355922564
// MI455X (gfx1250) — matrix operand order repaired and checked by reading, not compiled, not run on hardware
//
#include <hip/hip_runtime.h>

#define ALPHA 0.2f
#define NN 4096
#define EE 131072
#define DIN 512
#define DH 256
#define NHEAD 4

typedef __attribute__((ext_vector_type(16))) __bf16 v16bf;
typedef __attribute__((ext_vector_type(8)))  float  v8f;
typedef __attribute__((ext_vector_type(4)))  unsigned int v4u;
typedef __attribute__((ext_vector_type(8)))  int v8i;
typedef __attribute__((ext_vector_type(4)))  int v4i;
typedef __attribute__((address_space(3)))    __bf16 lds_bf16;

__device__ __forceinline__ float lrelu(float x) { return x > 0.f ? x : ALPHA * x; }

// ---------------- conversions / fills ----------------
__global__ void cvt_f32_to_bf16(const float* __restrict__ in, __bf16* __restrict__ out, int n) {
  int i = blockIdx.x * blockDim.x + threadIdx.x;
  if (i < n) out[i] = (__bf16)in[i];
}
__global__ void fill_f32(float* __restrict__ p, float v, int n) {
  int i = blockIdx.x * blockDim.x + threadIdx.x;
  if (i < n) p[i] = v;
}
__global__ void fill_i32(int* __restrict__ p, int v, int n) {
  int i = blockIdx.x * blockDim.x + threadIdx.x;
  if (i < n) p[i] = v;
}

// ---------------- WMMA bf16 GEMM: C[M,Nc] = A[M,K] * B[K,Nc] ----------------
// A fragment 16x32 (16-bit), CDNA5 layout: lanes 0-15 -> M, half selects K-block.
__device__ __forceinline__ v16bf load_a_frag(const __bf16* __restrict__ A, int lda,
                                             int row0, int k0, int lane) {
  int m = row0 + (lane & 15);
  int half = lane >> 4;
  const __bf16* p = A + (size_t)m * lda + k0 + half * 8;
  v16bf a;
#pragma unroll
  for (int e = 0; e < 16; ++e) {
    int pi = e >> 1, w = e & 1;
    int k = (pi < 4) ? (pi * 2 + w) : (16 + (pi - 4) * 2 + w);
    a[e] = p[k];
  }
  return a;
}
// B fragment 32x16 from LDS panel [K x 64]: lane&15 = N col, elems = K (table order, half selects K-block).
__device__ __forceinline__ v16bf load_b_frag_lds(const __bf16* __restrict__ ldsB,
                                                 int k0, int nlocal, int lane) {
  int half = lane >> 4;
  const __bf16* p = ldsB + (k0 + half * 8) * 64 + nlocal + (lane & 15);
  v16bf b;
#pragma unroll
  for (int e = 0; e < 16; ++e) b[e] = p[((e < 8) ? e : (e + 8)) * 64];
  return b;
}

// Block = 256 threads (8 waves). Block tile: 128 rows x 64 cols.
// Each wave: one 16-row strip, 4 column tiles (4 accumulators, A reused 4x).
// B panel (K x 64, <=64KB) staged in LDS via the Tensor Data Mover.
__global__ void __launch_bounds__(256)
gemm_bf16_wmma(const __bf16* __restrict__ A, int lda,
               const __bf16* __restrict__ B, int ldb, long strideB,
               float* __restrict__ C, long strideC,
               int M, int K, int Ncols) {
  __shared__ __bf16 ldsB[DIN * 64];  // max K = 512 -> 64KB
  int head = blockIdx.z;
  const __bf16* Bh = B + (size_t)head * strideB;
  float* Ch = C + (size_t)head * strideC;
  int n0 = blockIdx.y * 64;
  int row0 = blockIdx.x * 128 + (((int)threadIdx.x >> 5) << 4);
  int lane = threadIdx.x & 31;

#if __has_builtin(__builtin_amdgcn_tensor_load_to_lds)
  if (threadIdx.x < 32) {  // one wave issues the TDM descriptor (EXEC-independent DMA)
    unsigned long long gaddr = (unsigned long long)(const void*)(Bh + n0);
    unsigned lds_off = (unsigned)(unsigned long long)(lds_bf16*)&ldsB[0];
    v4u g0;
    g0[0] = 1u;                                   // count=1, user descriptor
    g0[1] = lds_off;                              // lds_addr (bytes)
    g0[2] = (unsigned)gaddr;                      // global_addr[31:0]
    g0[3] = (unsigned)((gaddr >> 32) & 0x1FFFFFFull) | 0x80000000u;  // addr[56:32] | type=2
    v8i g1;
    g1[0] = 0x00010000;                           // workgroup_mask=0, data_size=1 (2B)
    g1[1] = (int)(((unsigned)Ncols & 0xFFFFu) << 16);              // tensor_dim0 lo16
    g1[2] = (int)((((unsigned)Ncols >> 16) & 0xFFFFu) |
                  (((unsigned)K & 0xFFFFu) << 16));                // dim0 hi | tensor_dim1 lo
    g1[3] = (int)((((unsigned)K >> 16) & 0xFFFFu) | (64u << 16));  // dim1 hi | tile_dim0=64
    g1[4] = (int)((unsigned)K & 0xFFFFu);                          // tile_dim1=K, tile_dim2=0
    g1[5] = Ncols;                                                 // tensor_dim0_stride lo32
    g1[6] = 0; g1[7] = 0;                                          // stride hi / dim1_stride
    v4i g2 = {0, 0, 0, 0}, g3 = {0, 0, 0, 0};
#if defined(__clang_major__) && __clang_major__ >= 23
    v8i g4 = {0, 0, 0, 0, 0, 0, 0, 0};
    __builtin_amdgcn_tensor_load_to_lds(g0, g1, g2, g3, g4, 0);
#else
    __builtin_amdgcn_tensor_load_to_lds(g0, g1, g2, g3, 0);
#endif
    __builtin_amdgcn_s_wait_tensorcnt(0);
  }
  __syncthreads();
#else
  for (int idx = threadIdx.x; idx < K * 64; idx += 256)
    ldsB[idx] = Bh[(size_t)(idx >> 6) * ldb + n0 + (idx & 63)];
  __syncthreads();
#endif

  v8f c0 = {}, c1 = {}, c2 = {}, c3 = {};
  for (int k0 = 0; k0 < K; k0 += 32) {
    v16bf a  = load_a_frag(A, lda, row0, k0, lane);
    v16bf b0 = load_b_frag_lds(ldsB, k0, 0, lane);
    v16bf b1 = load_b_frag_lds(ldsB, k0, 16, lane);
    v16bf b2 = load_b_frag_lds(ldsB, k0, 32, lane);
    v16bf b3 = load_b_frag_lds(ldsB, k0, 48, lane);
    c0 = __builtin_amdgcn_wmma_f32_16x16x32_bf16(false, a, false, b0, (short)0, c0, false, false);
    c1 = __builtin_amdgcn_wmma_f32_16x16x32_bf16(false, a, false, b1, (short)0, c1, false, false);
    c2 = __builtin_amdgcn_wmma_f32_16x16x32_bf16(false, a, false, b2, (short)0, c2, false, false);
    c3 = __builtin_amdgcn_wmma_f32_16x16x32_bf16(false, a, false, b3, (short)0, c3, false, false);
  }
  // C/D layout: lane -> N = lane&15, VGPR r -> M = r + 8*(lane>>4)
  int col = n0 + (lane & 15);
  int rbase = row0 + ((lane >> 4) << 3);
#pragma unroll
  for (int r = 0; r < 8; ++r) {
    float* rowp = Ch + (size_t)(rbase + r) * Ncols + col;
    rowp[0]  = c0[r];
    rowp[16] = c1[r];
    rowp[32] = c2[r];
    rowp[48] = c3[r];
  }
}

// ---------------- attention pieces (factored dense-zero softmax) ----------------
__global__ void add_bias_kernel(float* __restrict__ prev, const float* __restrict__ pb) {
  int i = blockIdx.x, c = threadIdx.x;
  prev[(size_t)i * DH + c] += pb[c];
}

__global__ void wh_kernel(const float* __restrict__ hmat, const float* __restrict__ avec,
                          float* __restrict__ wh1, float* __restrict__ wh2) {
  int wid = blockIdx.x * (blockDim.x >> 5) + ((int)threadIdx.x >> 5);
  int lane = threadIdx.x & 31;
  if (wid >= NHEAD * NN) return;
  int head = wid / NN, i = wid % NN;
  const float* hp = hmat + ((size_t)head * NN + i) * DH;
  const float* a1 = avec + (size_t)head * 2 * DH;
  const float* a2 = a1 + DH;
  float s1 = 0.f, s2 = 0.f;
  for (int c = lane; c < DH; c += 32) { float hv = hp[c]; s1 += hv * a1[c]; s2 += hv * a2[c]; }
  for (int off = 16; off; off >>= 1) { s1 += __shfl_xor(s1, off, 32); s2 += __shfl_xor(s2, off, 32); }
  if (lane == 0) { wh1[wid] = s1; wh2[wid] = s2; }
}

__global__ void colsum_kernel(const float* __restrict__ hmat, float* __restrict__ colsum) {
  int head = blockIdx.x, c = threadIdx.x;
  const float* hp = hmat + (size_t)head * NN * DH;
  float s = 0.f;
  for (int i = 0; i < NN; ++i) s += hp[(size_t)i * DH + c];
  colsum[head * DH + c] = s;
}

__global__ void deg_kernel(const int* __restrict__ er, int* __restrict__ deg) {
  int e = blockIdx.x * blockDim.x + threadIdx.x;
  if (e < EE) atomicAdd(&deg[er[e]], 1);
}

__global__ void edge_max_kernel(const int* __restrict__ er, const int* __restrict__ ec,
                                const float* __restrict__ wh1, const float* __restrict__ wh2,
                                float* __restrict__ ev, float* __restrict__ rowmax) {
  int e = blockIdx.x * blockDim.x + threadIdx.x;
  int head = blockIdx.y;
  if (e >= EE) return;
  int i = er[e], j = ec[e];
  float s = lrelu(wh1[head * NN + i] + wh2[head * NN + j]);
  ev[(size_t)head * EE + e] = s;
  // zeros participate in softmax -> rowmax starts at 0; positive floats order as uints
  if (s > 0.f) atomicMax((unsigned int*)&rowmax[head * NN + i], __float_as_uint(s));
}

__global__ void edge_sum_kernel(const int* __restrict__ er, const float* __restrict__ ev,
                                const float* __restrict__ rowmax, float* __restrict__ rowsum) {
  int e = blockIdx.x * blockDim.x + threadIdx.x;
  int head = blockIdx.y;
  if (e >= EE) return;
  int hi = head * NN + er[e];
  atomicAdd(&rowsum[hi], __expf(ev[(size_t)head * EE + e] - rowmax[hi]));
}

__global__ void stats_kernel(const float* __restrict__ rowmax, const float* __restrict__ rowsum,
                             const int* __restrict__ deg, float* __restrict__ Z,
                             float* __restrict__ base) {
  int idx = blockIdx.x * blockDim.x + threadIdx.x;
  if (idx >= NHEAD * NN) return;
  int i = idx % NN;
  float m = rowmax[idx];
  float em = __expf(-m);
  float z = rowsum[idx] + (float)(NN - deg[i]) * em;  // (N-deg) zero-entries contribute exp(-m)
  Z[idx] = z;
  base[idx] = em / z;
}

__global__ void agg_init_kernel(const float* __restrict__ base, const float* __restrict__ colsum,
                                float* __restrict__ agg) {
  int i = blockIdx.x, head = blockIdx.y, c = threadIdx.x;
  agg[((size_t)head * NN + i) * DH + c] = base[head * NN + i] * colsum[head * DH + c];
}

__global__ void edge_agg_kernel(const int* __restrict__ er, const int* __restrict__ ec,
                                const float* __restrict__ ev, const float* __restrict__ rowmax,
                                const float* __restrict__ Z, const float* __restrict__ base,
                                const float* __restrict__ hmat, float* __restrict__ agg) {
  int e = blockIdx.x, head = blockIdx.y, c = threadIdx.x;
  int i = er[e], j = ec[e];
  int hi = head * NN + i;
  float coef = __expf(ev[(size_t)head * EE + e] - rowmax[hi]) / Z[hi] - base[hi];
  atomicAdd(&agg[((size_t)head * NN + i) * DH + c],
            coef * hmat[((size_t)head * NN + j) * DH + c]);
}

// leaky_relu -> L2 normalize per head -> +bias -> mean heads -> +prev -> ELU
__global__ void finalize_kernel(const float* __restrict__ agg, const float* __restrict__ bias,
                                float* __restrict__ prev, __bf16* __restrict__ xb16,
                                float* __restrict__ dout, int write_out) {
  __shared__ float red[DH];
  int i = blockIdx.x, c = threadIdx.x;
  float acc = 0.f;
  for (int head = 0; head < NHEAD; ++head) {
    float v = lrelu(agg[((size_t)head * NN + i) * DH + c]);
    red[c] = v * v;
    __syncthreads();
    for (int s = DH / 2; s > 0; s >>= 1) { if (c < s) red[c] += red[c + s]; __syncthreads(); }
    float nrm = sqrtf(red[0]);
    __syncthreads();
    acc += v / fmaxf(nrm, 1e-12f) + bias[head * DH + c];
  }
  float o = acc * (1.f / NHEAD) + prev[(size_t)i * DH + c];
  o = o > 0.f ? o : (__expf(o) - 1.f);
  prev[(size_t)i * DH + c] = o;            // residual for next block
  xb16[(size_t)i * DH + c] = (__bf16)o;    // bf16 input for next block's GEMM
  if (write_out) dout[(size_t)i * DH + c] = o;
}

extern "C" void kernel_launch(void* const* d_in, const int* in_sizes, int n_in,
                              void* d_out, int out_size, void* d_ws, size_t ws_size,
                              hipStream_t stream) {
  (void)in_sizes; (void)n_in; (void)out_size; (void)ws_size;
  const float* x  = (const float*)d_in[0];
  const int* edge = (const int*)d_in[1];
  const int* er = edge;        // edge_index[0]
  const int* ec = edge + EE;   // edge_index[1]
  const float* W1 = (const float*)d_in[2];
  const float* a1 = (const float*)d_in[3];
  const float* b1 = (const float*)d_in[4];
  const float* Wk = (const float*)d_in[5];
  const float* ak = (const float*)d_in[6];
  const float* bk = (const float*)d_in[7];
  const float* pW = (const float*)d_in[8];
  const float* pb = (const float*)d_in[9];
  float* dout = (float*)d_out;

  char* ws = (char*)d_ws;
  size_t off = 0;
  auto alloc = [&](size_t bytes) -> void* {
    void* p = ws + off; off += (bytes + 255) & ~(size_t)255; return p;
  };
  __bf16* xb    = (__bf16*)alloc(sizeof(__bf16) * NN * DIN);
  __bf16* wb    = (__bf16*)alloc(sizeof(__bf16) * NHEAD * DIN * DH);
  __bf16* pwb   = (__bf16*)alloc(sizeof(__bf16) * DIN * DH);
  float* hmat   = (float*)alloc(sizeof(float) * NHEAD * NN * DH);
  float* agg    = (float*)alloc(sizeof(float) * NHEAD * NN * DH);
  float* wh1    = (float*)alloc(sizeof(float) * NHEAD * NN);
  float* wh2    = (float*)alloc(sizeof(float) * NHEAD * NN);
  float* rowmax = (float*)alloc(sizeof(float) * NHEAD * NN);
  float* rowsum = (float*)alloc(sizeof(float) * NHEAD * NN);
  float* Zb     = (float*)alloc(sizeof(float) * NHEAD * NN);
  float* baseb  = (float*)alloc(sizeof(float) * NHEAD * NN);
  float* colsum = (float*)alloc(sizeof(float) * NHEAD * DH);
  float* evb    = (float*)alloc(sizeof(float) * NHEAD * EE);
  int*   deg    = (int*)alloc(sizeof(int) * NN);
  float* prev   = (float*)alloc(sizeof(float) * NN * DH);

  // x -> bf16; prev = x @ pW + pb (WMMA)
  cvt_f32_to_bf16<<<(NN * DIN + 255) / 256, 256, 0, stream>>>(x, xb, NN * DIN);
  cvt_f32_to_bf16<<<(DIN * DH + 255) / 256, 256, 0, stream>>>(pW, pwb, DIN * DH);
  gemm_bf16_wmma<<<dim3(NN / 128, DH / 64, 1), 256, 0, stream>>>(
      xb, DIN, pwb, DH, 0, prev, 0, NN, DIN, DH);
  add_bias_kernel<<<NN, DH, 0, stream>>>(prev, pb);

  // degree is layer-invariant
  fill_i32<<<(NN + 255) / 256, 256, 0, stream>>>(deg, 0, NN);
  deg_kernel<<<(EE + 255) / 256, 256, 0, stream>>>(er, deg);

  for (int l = 0; l < 4; ++l) {
    int K = (l == 0) ? DIN : DH;
    const float* Wsrc = (l == 0) ? W1 : Wk + (size_t)(l - 1) * NHEAD * DH * DH;
    const float* asrc = (l == 0) ? a1 : ak + (size_t)(l - 1) * NHEAD * 2 * DH;
    const float* bsrc = (l == 0) ? b1 : bk + (size_t)(l - 1) * NHEAD * DH;

    int wcnt = NHEAD * K * DH;
    cvt_f32_to_bf16<<<(wcnt + 255) / 256, 256, 0, stream>>>(Wsrc, wb, wcnt);
    gemm_bf16_wmma<<<dim3(NN / 128, DH / 64, NHEAD), 256, 0, stream>>>(
        xb, K, wb, DH, (long)K * DH, hmat, (long)NN * DH, NN, K, DH);

    colsum_kernel<<<NHEAD, DH, 0, stream>>>(hmat, colsum);
    wh_kernel<<<(NHEAD * NN * 32 + 127) / 128, 128, 0, stream>>>(hmat, asrc, wh1, wh2);

    fill_f32<<<(NHEAD * NN + 255) / 256, 256, 0, stream>>>(rowmax, 0.f, NHEAD * NN);
    fill_f32<<<(NHEAD * NN + 255) / 256, 256, 0, stream>>>(rowsum, 0.f, NHEAD * NN);
    dim3 eg((EE + 255) / 256, NHEAD);
    edge_max_kernel<<<eg, 256, 0, stream>>>(er, ec, wh1, wh2, evb, rowmax);
    edge_sum_kernel<<<eg, 256, 0, stream>>>(er, evb, rowmax, rowsum);
    stats_kernel<<<(NHEAD * NN + 255) / 256, 256, 0, stream>>>(rowmax, rowsum, deg, Zb, baseb);

    agg_init_kernel<<<dim3(NN, NHEAD), DH, 0, stream>>>(baseb, colsum, agg);
    edge_agg_kernel<<<dim3(EE, NHEAD), DH, 0, stream>>>(er, ec, evb, rowmax, Zb, baseb,
                                                        hmat, agg);
    finalize_kernel<<<NN, DH, 0, stream>>>(agg, bsrc, prev, xb, dout, (l == 3) ? 1 : 0);
  }
}
